// ImplicitMoE_68161130987768
// MI455X (gfx1250) — compile-verified, matrix operand order checked by reading, not run on hardware
//
#include <hip/hip_runtime.h>
#include <hip/hip_bf16.h>

// Problem constants (match reference)
#define N_NEURONS     8192
#define NUM_GROUPS    64
#define NPG           128   // neurons per group
#define ROWS_PER_TILE 16    // batch rows handled per wave (WMMA N dim)
#define WAVES_PER_BLK 8     // 256 threads, wave32

typedef __attribute__((ext_vector_type(16))) __bf16 v16bf;
typedef __attribute__((ext_vector_type(8)))  float  v8f;
typedef __attribute__((ext_vector_type(4)))  float  f32x4;

// One wave computes a 16-row x 128-col (one group) tile.
// Lane l owns batch row (l & 15); lanes l and l+16 split the 128 group
// columns (64 f32 each). Data goes into the WMMA *B* operand; A is all-ones,
// so C[m][n] = sum_k B[k][n] = group-sum of row n, and since C's layout puts
// column n = lane&15 in every VGPR, each lane reads its own row sum from c[0].
// f32 precision is preserved via bf16 hi/lo split (two accumulating WMMAs
// per 32-wide K chunk).
__global__ __launch_bounds__(256, 1)
void group_inhibition_kernel(const float* __restrict__ spk,
                             float* __restrict__ out)
{
    const int lane = threadIdx.x & 31;
    const int wave = threadIdx.x >> 5;
    const int tile = blockIdx.x * WAVES_PER_BLK + wave;

    const int g  = tile & (NUM_GROUPS - 1);   // group index
    const int rt = tile >> 6;                 // row-tile index
    const int row     = rt * ROWS_PER_TILE + (lane & 15);
    const int halfsel = lane >> 4;            // which 16-col half per chunk

    const size_t base = (size_t)row * N_NEURONS + (size_t)g * NPG
                      + (size_t)halfsel * 16;
    const float* __restrict__ p = spk + base;

    // Load this lane's 64 f32 values (4 K-chunks x 16 values) as b128 loads.
    f32x4 x[16];
#pragma unroll
    for (int i = 0; i < 4; ++i)
#pragma unroll
        for (int j = 0; j < 4; ++j)
            x[i * 4 + j] = *(const f32x4*)(p + i * 32 + j * 4);

    // A operand: all ones (layout-invariant).
    v16bf ones;
#pragma unroll
    for (int e = 0; e < 16; ++e) ones[e] = (__bf16)1.0f;

    v8f c = {};  // f32 accumulator, starts at zero
#pragma unroll
    for (int i = 0; i < 4; ++i) {
        v16bf bhi, blo;
#pragma unroll
        for (int q = 0; q < 4; ++q) {
#pragma unroll
            for (int e = 0; e < 4; ++e) {
                const float xv = x[i * 4 + q][e];
                const __bf16 h = (__bf16)xv;          // top ~8 mantissa bits
                bhi[q * 4 + e] = h;
                blo[q * 4 + e] = (__bf16)(xv - (float)h); // exact residual, rounded
            }
        }
        // (neg_a, A, neg_b, B, c_mod, C, reuse_a, reuse_b)
        c = __builtin_amdgcn_wmma_f32_16x16x32_bf16(
                false, ones, false, bhi, (short)0, c, false, false);
        c = __builtin_amdgcn_wmma_f32_16x16x32_bf16(
                false, ones, false, blo, (short)0, c, false, false);
    }

    // Every VGPR of C holds this lane's column (= its batch row) group sum.
    const float half_sum = 0.5f * c[0];   // exact scaling by 0.5

    float* __restrict__ qo = out + base;
#pragma unroll
    for (int i = 0; i < 4; ++i)
#pragma unroll
        for (int j = 0; j < 4; ++j) {
            const f32x4 v = x[i * 4 + j];
            f32x4 r;
#pragma unroll
            for (int e = 0; e < 4; ++e)
                r[e] = __builtin_fmaf(-0.5f, v[e], half_sum); // == (sum-x)*0.5
            // Non-temporal store: keep the 128MB input resident in 192MB L2
            // across graph replays instead of letting output thrash it.
            __builtin_nontemporal_store(r, (f32x4*)(qo + i * 32 + j * 4));
        }
}

extern "C" void kernel_launch(void* const* d_in, const int* in_sizes, int n_in,
                              void* d_out, int out_size, void* d_ws, size_t ws_size,
                              hipStream_t stream) {
    (void)n_in; (void)d_ws; (void)ws_size; (void)out_size;
    const float* spk = (const float*)d_in[0];
    float* out = (float*)d_out;

    const int batch = in_sizes[0] / N_NEURONS;                 // 4096
    const int tiles = (batch / ROWS_PER_TILE) * NUM_GROUPS;    // 16384 waves
    const int blocks = tiles / WAVES_PER_BLK;                  // 2048

    hipLaunchKernelGGL(group_inhibition_kernel,
                       dim3(blocks), dim3(256), 0, stream, spk, out);
}
